// LSTMdecoder_64665027608597
// MI455X (gfx1250) — matrix operand order repaired and checked by reading, not compiled, not run on hardware
//
#include <hip/hip_runtime.h>
#include <hip/hip_bf16.h>

// ---------------------------------------------------------------------------
// MI455X (gfx1250) attention + 2-layer LSTM decoder.
//  * All GEMMs on v_wmma_f32_16x16x32_bf16 (bf16 in / f32 accumulate).
//  * Register-blocked wave tiles (2Mx4N) => 8 WMMA per 6 fragment loads.
//  * B operands pre-transposed to [N][K] so fragments are contiguous b128s.
//  * LSTM input projection hoisted out of the recurrence (one 8192x2048x512).
//  * Recurrent h staged into LDS via TENSOR_LOAD_TO_LDS (TDM) + tensorcnt.
// ---------------------------------------------------------------------------

typedef __attribute__((ext_vector_type(16))) __bf16 bf16x16;
typedef __attribute__((ext_vector_type(8)))  __bf16 bf16x8;
typedef __attribute__((ext_vector_type(8)))  float  f32x8;
typedef __attribute__((ext_vector_type(4)))  unsigned int u32x4;
typedef __attribute__((ext_vector_type(4)))  int i32x4;
typedef __attribute__((ext_vector_type(8)))  int i32x8;
typedef long long i64;

#define B_SZ    32
#define S_ENC   512
#define S_DEC   256
#define DM      512      // d_model == SIZE
#define G4      2048     // 4*SIZE
#define MROWS   (B_SZ * S_DEC)   // 8192

__device__ __forceinline__ __bf16 f2bf(float f) {
  unsigned u = __float_as_uint(f);
  unsigned r = (u + 0x7FFFu + ((u >> 16) & 1u)) >> 16;
  unsigned short s = (unsigned short)r;
  return __builtin_bit_cast(__bf16, s);
}

__device__ __forceinline__ float sigm(float x) {
  return 1.0f / (1.0f + __expf(-x));
}

__device__ __forceinline__ bf16x16 join16(bf16x8 lo, bf16x8 hi) {
  return __builtin_shufflevector(lo, hi, 0,1,2,3,4,5,6,7,8,9,10,11,12,13,14,15);
}

// ---- Tensor Data Mover: 1-D DMA of `dwords` dwords global -> LDS ----------
// D# per CDNA5 ISA 8.3/8.4: group0 {count=1, lds_addr, global_addr, type=2},
// group1 {data_size=4B, tensor_dim0=tile_dim0=dwords, dim0_stride=dwords}.
__device__ __forceinline__ void tdm_load_to_lds(void* lds_ptr, const void* gptr,
                                                unsigned dwords) {
  unsigned lds_off = (unsigned)(size_t)lds_ptr;          // generic low 32 = LDS addr
  unsigned long long ga = (unsigned long long)(size_t)gptr;
  u32x4 g0;
  g0[0] = 1u;                                            // count=1 (user D#)
  g0[1] = lds_off;                                       // lds_addr
  g0[2] = (unsigned)(ga & 0xFFFFFFFFu);                  // global_addr[31:0]
  g0[3] = (unsigned)((ga >> 32) & 0x1FFFFFFu) | (2u << 30); // addr[56:32], type=2
  i32x8 g1;
  g1[0] = (int)(2u << 16);                               // data_size = 4 bytes
  g1[1] = (int)((dwords & 0xFFFFu) << 16);               // tensor_dim0[15:0]
  g1[2] = (int)(((dwords >> 16) & 0xFFFFu) | (1u << 16));// dim0[31:16], tensor_dim1=1
  g1[3] = (int)((dwords & 0xFFFFu) << 16);               // tile_dim0
  g1[4] = 0;                                             // tile_dim1/2 unused
  g1[5] = (int)dwords;                                   // tensor_dim0_stride[31:0]
  g1[6] = 0;
  g1[7] = 0;
  i32x4 z4; z4[0] = 0; z4[1] = 0; z4[2] = 0; z4[3] = 0;
#if __clang_major__ >= 23
  i32x8 z8; for (int i = 0; i < 8; ++i) z8[i] = 0;
  __builtin_amdgcn_tensor_load_to_lds(g0, g1, z4, z4, z8, 0);
#else
  __builtin_amdgcn_tensor_load_to_lds(g0, g1, z4, z4, 0);
#endif
}

// --------------------------- elementwise helpers ---------------------------

__global__ void convert_f32_to_bf16(const float* __restrict__ in,
                                    __bf16* __restrict__ out, i64 n) {
  i64 i = (i64)blockIdx.x * blockDim.x + threadIdx.x;
  if (i < n) out[i] = f2bf(in[i]);
}

// in: f32 [batch][R][C]  ->  out: bf16 [batch][C][R]
__global__ void transpose_f32_to_bf16(const float* __restrict__ in,
                                      __bf16* __restrict__ out,
                                      int R, int C, i64 inB, i64 outB) {
  i64 bi = (i64)blockIdx.z * inB;
  i64 bo = (i64)blockIdx.z * outB;
  int idx = blockIdx.x * blockDim.x + threadIdx.x;
  if (idx < R * C) {
    int r = idx / C, c = idx % C;
    out[bo + (i64)c * R + r] = f2bf(in[bi + idx]);
  }
}

// y[m][k] = bf16(emb[last_word[m]][k]),  m in [0, 8192)
__global__ void gather_embed(const int* __restrict__ words,
                             const float* __restrict__ emb,
                             __bf16* __restrict__ y, int total) {
  int i = blockIdx.x * blockDim.x + threadIdx.x;
  if (i < total) {
    int m = i >> 9;
    int k = i & 511;
    int w = words[m];
    y[i] = f2bf(emb[(i64)w * DM + k]);
  }
}

__global__ void lstm_init(float* __restrict__ c0, float* __restrict__ c1,
                          __bf16* __restrict__ h0, __bf16* __restrict__ h1) {
  int i = blockIdx.x * blockDim.x + threadIdx.x;
  if (i < B_SZ * DM) {
    c0[i] = 0.0f; c1[i] = 0.0f;
    h0[i] = f2bf(0.0f); h1[i] = f2bf(0.0f);
  }
}

// ------------------------------ generic GEMM -------------------------------
// C[M][N] = A[M][K] * BT[N][K]^T (+bias[n]).
// Workgroup: 256 threads = 8 waves as 2(wm) x 4(wn); each wave register-blocks
// a 32x64 tile (2 M-subtiles x 4 N-subtiles, 8 f32x8 accumulators).
// Workgroup tile: 64M x 256N.  M = gridDim.y*64, N = gridDim.x*256, K % 32 == 0.
__global__ __launch_bounds__(256)
void gemm_bf16_wmma(const __bf16* __restrict__ A, i64 aBatch,
                    const __bf16* __restrict__ BT, i64 bBatch,
                    float* __restrict__ Cf, __bf16* __restrict__ Cb, i64 cBatch,
                    const float* __restrict__ bias, int N, int K) {
  const int lane = threadIdx.x & 31;
  const int w    = threadIdx.x >> 5;
  const int wm   = w >> 2, wn = w & 3;
  const int m0   = blockIdx.y * 64 + wm * 32;
  const int n0   = blockIdx.x * 256 + wn * 64;
  const int r    = lane & 15;
  const int hi   = lane >> 4;

  const __bf16* aB = A  + (i64)blockIdx.z * aBatch;
  const __bf16* bB = BT + (i64)blockIdx.z * bBatch;
  const __bf16* ap[2];
  const __bf16* bp[4];
  #pragma unroll
  for (int i = 0; i < 2; ++i) ap[i] = aB + (i64)(m0 + 16 * i + r) * K + hi * 8;
  #pragma unroll
  for (int j = 0; j < 4; ++j) bp[j] = bB + (i64)(n0 + 16 * j + r) * K + hi * 8;

  f32x8 acc[2][4];
  #pragma unroll
  for (int i = 0; i < 2; ++i)
    #pragma unroll
    for (int j = 0; j < 4; ++j)
      #pragma unroll
      for (int v = 0; v < 8; ++v) acc[i][j][v] = 0.0f;

  #pragma unroll 2
  for (int kk = 0; kk < K; kk += 32) {
    bf16x16 af[2], bfr[4];
    #pragma unroll
    for (int i = 0; i < 2; ++i) {
      __builtin_prefetch(ap[i] + kk + 128, 0, 1);
      af[i] = join16(*(const bf16x8*)(ap[i] + kk),
                     *(const bf16x8*)(ap[i] + kk + 16));
    }
    #pragma unroll
    for (int j = 0; j < 4; ++j) {
      __builtin_prefetch(bp[j] + kk + 128, 0, 1);
      bfr[j] = join16(*(const bf16x8*)(bp[j] + kk),
                      *(const bf16x8*)(bp[j] + kk + 16));
    }
    #pragma unroll
    for (int i = 0; i < 2; ++i)
      #pragma unroll
      for (int j = 0; j < 4; ++j)
        acc[i][j] = __builtin_amdgcn_wmma_f32_16x16x32_bf16(
            false, af[i], false, bfr[j], (short)0, acc[i][j], false, false);
  }

  const i64 cb = (i64)blockIdx.z * cBatch;
  #pragma unroll
  for (int j = 0; j < 4; ++j) {
    const int cn = n0 + 16 * j + r;
    const float bv = bias ? bias[cn] : 0.0f;
    #pragma unroll
    for (int i = 0; i < 2; ++i) {
      #pragma unroll
      for (int v = 0; v < 8; ++v) {
        int row = m0 + 16 * i + hi * 8 + v;
        float val = acc[i][j][v] + bv;
        i64 ci = cb + (i64)row * N + cn;
        if (Cf) Cf[ci] = val;
        if (Cb) Cb[ci] = f2bf(val);
      }
    }
  }
}

// ---------------------- softmax over encoder axis --------------------------
// scores[b][e][s] f32 + mask[b][e][s] int -> at[b][s][e] bf16, softmax over e.
__global__ __launch_bounds__(256)
void softmax_cols(const float* __restrict__ scores, const int* __restrict__ mask,
                  __bf16* __restrict__ at) {
  int gw   = (blockIdx.x * 256 + threadIdx.x) >> 5;   // 0 .. 8191
  int lane = threadIdx.x & 31;
  int b = gw >> 8, s = gw & 255;
  const float* col = scores + (i64)b * S_ENC * S_DEC + s;
  const int*   mk  = mask   + (i64)b * S_ENC * S_DEC + s;

  float v[16];
  float mx = -1e30f;
  #pragma unroll
  for (int i = 0; i < 16; ++i) {
    int e = i * 32 + lane;
    float x = col[(i64)e * S_DEC];
    if (mk[(i64)e * S_DEC]) x -= 1e9f;
    v[i] = x;
    mx = fmaxf(mx, x);
  }
  #pragma unroll
  for (int off = 16; off > 0; off >>= 1) mx = fmaxf(mx, __shfl_xor(mx, off, 32));
  float sum = 0.0f;
  #pragma unroll
  for (int i = 0; i < 16; ++i) { v[i] = __expf(v[i] - mx); sum += v[i]; }
  #pragma unroll
  for (int off = 16; off > 0; off >>= 1) sum += __shfl_xor(sum, off, 32);
  float inv = 1.0f / sum;
  __bf16* orow = at + ((i64)b * S_DEC + s) * S_ENC;
  #pragma unroll
  for (int i = 0; i < 16; ++i) orow[i * 32 + lane] = f2bf(v[i] * inv);
}

// ------------------------------ LSTM steps ---------------------------------
#define TJ 32          // hidden columns per workgroup
#define NT (TJ / 16)

// Layer 0: z = X0[:,t,:] + h_in @ r0T^T ; gate ; update c, h_out.
__global__ __launch_bounds__(256)
void lstm_layer0(const float* __restrict__ X0, int t,
                 const __bf16* __restrict__ h_in, __bf16* __restrict__ h_out,
                 float* __restrict__ c, const __bf16* __restrict__ r0T) {
  __shared__ __attribute__((aligned(16))) __bf16 lds_h[B_SZ * DM];   // 32 KB
  __shared__ float lds_z[4][B_SZ * TJ];                              // 16 KB

  if ((threadIdx.x >> 5) == 0) {        // wave 0 issues the DMA (EXEC ignored)
    tdm_load_to_lds(&lds_h[0], h_in, (B_SZ * DM * 2) / 4);
    __builtin_amdgcn_s_wait_tensorcnt(0);
  }
  __syncthreads();

  const int lane = threadIdx.x & 31;
  const int w    = threadIdx.x >> 5;
  const int g    = w >> 1;        // gate 0..3 (i,f,g,o)
  const int m0   = (w & 1) * 16;
  const int r    = lane & 15;
  const int hi   = lane >> 4;
  const int j0   = blockIdx.x * TJ;

  const __bf16* ap = lds_h + (m0 + r) * DM + hi * 8;
  const __bf16* bp[NT];
  #pragma unroll
  for (int nt = 0; nt < NT; ++nt)
    bp[nt] = r0T + (i64)(g * DM + j0 + nt * 16 + r) * DM + hi * 8;

  f32x8 acc[NT];
  #pragma unroll
  for (int nt = 0; nt < NT; ++nt)
    #pragma unroll
    for (int i = 0; i < 8; ++i) acc[nt][i] = 0.0f;

  #pragma unroll 4
  for (int kk = 0; kk < DM; kk += 32) {
    bf16x16 af = join16(*(const bf16x8*)(ap + kk),
                        *(const bf16x8*)(ap + kk + 16));   // A reused across NT
    #pragma unroll
    for (int nt = 0; nt < NT; ++nt) {
      bf16x16 bfr = join16(*(const bf16x8*)(bp[nt] + kk),
                           *(const bf16x8*)(bp[nt] + kk + 16));
      acc[nt] = __builtin_amdgcn_wmma_f32_16x16x32_bf16(
          false, af, false, bfr, (short)0, acc[nt], false, false);
    }
  }

  #pragma unroll
  for (int nt = 0; nt < NT; ++nt) {
    const int ncol = g * DM + j0 + nt * 16 + r;
    #pragma unroll
    for (int v = 0; v < 8; ++v) {
      int m = m0 + hi * 8 + v;     // batch index
      float z = acc[nt][v] + X0[((i64)m * S_DEC + t) * G4 + ncol];
      lds_z[g][m * TJ + nt * 16 + r] = z;
    }
  }
  __syncthreads();

  for (int i = threadIdx.x; i < B_SZ * TJ; i += 256) {
    int m = i / TJ, jl = i % TJ;
    int j = j0 + jl;
    float zi = lds_z[0][i], zf = lds_z[1][i], zg = lds_z[2][i], zo = lds_z[3][i];
    float cold = c[m * DM + j];
    float cn = sigm(zf) * cold + sigm(zi) * tanhf(zg);
    float hn = sigm(zo) * tanhf(cn);
    c[m * DM + j] = cn;
    h_out[m * DM + j] = f2bf(hn);
  }
}

// Layer 1: z = h0 @ k1T^T + h1_in @ r1T^T + b1 ; gate ; out[b][t][:] & h1_out.
__global__ __launch_bounds__(256)
void lstm_layer1(int t, const __bf16* __restrict__ h0,
                 const __bf16* __restrict__ h1_in, __bf16* __restrict__ h1_out,
                 float* __restrict__ c, const __bf16* __restrict__ k1T,
                 const __bf16* __restrict__ r1T, const float* __restrict__ b1,
                 float* __restrict__ out) {
  __shared__ __attribute__((aligned(16))) __bf16 lds_h0[B_SZ * DM];  // 32 KB
  __shared__ float lds_z[4][B_SZ * TJ];                              // 16 KB

  if ((threadIdx.x >> 5) == 0) {
    tdm_load_to_lds(&lds_h0[0], h0, (B_SZ * DM * 2) / 4);
    __builtin_amdgcn_s_wait_tensorcnt(0);
  }
  __syncthreads();

  const int lane = threadIdx.x & 31;
  const int w    = threadIdx.x >> 5;
  const int g    = w >> 1;
  const int m0   = (w & 1) * 16;
  const int r    = lane & 15;
  const int hi   = lane >> 4;
  const int j0   = blockIdx.x * TJ;

  const __bf16* ap0 = lds_h0 + (m0 + r) * DM + hi * 8;
  const __bf16* ap1 = h1_in + (i64)(m0 + r) * DM + hi * 8;
  const __bf16* bpk[NT];
  const __bf16* bpr[NT];
  #pragma unroll
  for (int nt = 0; nt < NT; ++nt) {
    i64 off = (i64)(g * DM + j0 + nt * 16 + r) * DM + hi * 8;
    bpk[nt] = k1T + off;
    bpr[nt] = r1T + off;
  }

  f32x8 acc[NT];
  #pragma unroll
  for (int nt = 0; nt < NT; ++nt)
    #pragma unroll
    for (int i = 0; i < 8; ++i) acc[nt][i] = 0.0f;

  #pragma unroll 4
  for (int kk = 0; kk < DM; kk += 32) {
    bf16x16 af = join16(*(const bf16x8*)(ap0 + kk),
                        *(const bf16x8*)(ap0 + kk + 16));
    #pragma unroll
    for (int nt = 0; nt < NT; ++nt) {
      bf16x16 bfr = join16(*(const bf16x8*)(bpk[nt] + kk),
                           *(const bf16x8*)(bpk[nt] + kk + 16));
      acc[nt] = __builtin_amdgcn_wmma_f32_16x16x32_bf16(
          false, af, false, bfr, (short)0, acc[nt], false, false);
    }
  }
  #pragma unroll 4
  for (int kk = 0; kk < DM; kk += 32) {
    bf16x16 af = join16(*(const bf16x8*)(ap1 + kk),
                        *(const bf16x8*)(ap1 + kk + 16));
    #pragma unroll
    for (int nt = 0; nt < NT; ++nt) {
      bf16x16 bfr = join16(*(const bf16x8*)(bpr[nt] + kk),
                           *(const bf16x8*)(bpr[nt] + kk + 16));
      acc[nt] = __builtin_amdgcn_wmma_f32_16x16x32_bf16(
          false, af, false, bfr, (short)0, acc[nt], false, false);
    }
  }

  #pragma unroll
  for (int nt = 0; nt < NT; ++nt) {
    const int ncol = g * DM + j0 + nt * 16 + r;
    #pragma unroll
    for (int v = 0; v < 8; ++v) {
      int m = m0 + hi * 8 + v;
      lds_z[g][m * TJ + nt * 16 + r] = acc[nt][v] + b1[ncol];
    }
  }
  __syncthreads();

  for (int i = threadIdx.x; i < B_SZ * TJ; i += 256) {
    int m = i / TJ, jl = i % TJ;
    int j = j0 + jl;
    float zi = lds_z[0][i], zf = lds_z[1][i], zg = lds_z[2][i], zo = lds_z[3][i];
    float cold = c[m * DM + j];
    float cn = sigm(zf) * cold + sigm(zi) * tanhf(zg);
    float hn = sigm(zo) * tanhf(cn);
    c[m * DM + j] = cn;
    h1_out[m * DM + j] = f2bf(hn);
    out[((i64)m * S_DEC + t) * DM + j] = hn;
  }
}

// ------------------------------- launcher ----------------------------------

extern "C" void kernel_launch(void* const* d_in, const int* in_sizes, int n_in,
                              void* d_out, int out_size, void* d_ws, size_t ws_size,
                              hipStream_t stream) {
  const float* enc   = (const float*)d_in[0];
  const int*   words = (const int*)  d_in[1];
  const int*   mask  = (const int*)  d_in[2];
  const float* P_w   = (const float*)d_in[3];
  const float* P_b   = (const float*)d_in[4];
  const float* emb   = (const float*)d_in[5];
  const float* din_w = (const float*)d_in[6];
  const float* din_b = (const float*)d_in[7];
  const float* k0    = (const float*)d_in[8];
  const float* r0    = (const float*)d_in[9];
  const float* b0    = (const float*)d_in[10];
  const float* k1    = (const float*)d_in[11];
  const float* r1    = (const float*)d_in[12];
  const float* b1    = (const float*)d_in[13];
  float* out = (float*)d_out;

  char* ws = (char*)d_ws;
  auto alloc = [&](size_t bytes) -> char* {
    char* p = ws; ws += (bytes + 255) & ~(size_t)255; return p;
  };

  __bf16* enc_bf  = (__bf16*)alloc((i64)B_SZ * S_ENC * DM * 2);  // [b][e][d]
  __bf16* encT_bf = (__bf16*)alloc((i64)B_SZ * DM * S_ENC * 2);  // [b][d][e]
  __bf16* PwT     = (__bf16*)alloc((i64)DM * DM * 2);
  __bf16* dinT    = (__bf16*)alloc((i64)DM * DM * 2);
  __bf16* k0T     = (__bf16*)alloc((i64)G4 * DM * 2);
  __bf16* r0T     = (__bf16*)alloc((i64)G4 * DM * 2);
  __bf16* k1T     = (__bf16*)alloc((i64)G4 * DM * 2);
  __bf16* r1T     = (__bf16*)alloc((i64)G4 * DM * 2);
  __bf16* y_bf    = (__bf16*)alloc((i64)MROWS * DM * 2);
  __bf16* py_bf   = (__bf16*)alloc((i64)MROWS * DM * 2);         // [b][s][d]
  float*  scores  = (float*) alloc((i64)B_SZ * S_ENC * S_DEC * 4);
  __bf16* at_bf   = (__bf16*)alloc((i64)B_SZ * S_DEC * S_ENC * 2); // [b][s][e]
  __bf16* dec_bf  = (__bf16*)alloc((i64)MROWS * DM * 2);
  __bf16* x_bf    = (__bf16*)alloc((i64)MROWS * DM * 2);
  float*  X0      = (float*) alloc((i64)MROWS * G4 * 4);
  __bf16* h0a     = (__bf16*)alloc((i64)B_SZ * DM * 2);
  __bf16* h0b     = (__bf16*)alloc((i64)B_SZ * DM * 2);
  __bf16* h1a     = (__bf16*)alloc((i64)B_SZ * DM * 2);
  __bf16* h1b     = (__bf16*)alloc((i64)B_SZ * DM * 2);
  float*  c0      = (float*) alloc((i64)B_SZ * DM * 4);
  float*  c1      = (float*) alloc((i64)B_SZ * DM * 4);
  (void)ws_size; (void)in_sizes; (void)n_in; (void)out_size;

  // ---- precision conversions / transposes ----
  {
    i64 nEnc = (i64)B_SZ * S_ENC * DM;
    convert_f32_to_bf16<<<(int)((nEnc + 255) / 256), 256, 0, stream>>>(enc, enc_bf, nEnc);
    dim3 gE((S_ENC * DM + 255) / 256, 1, B_SZ);
    transpose_f32_to_bf16<<<gE, 256, 0, stream>>>(enc, encT_bf, S_ENC, DM,
                                                  (i64)S_ENC * DM, (i64)DM * S_ENC);
    transpose_f32_to_bf16<<<dim3((DM * DM + 255) / 256, 1, 1), 256, 0, stream>>>(
        P_w, PwT, DM, DM, 0, 0);
    transpose_f32_to_bf16<<<dim3((DM * DM + 255) / 256, 1, 1), 256, 0, stream>>>(
        din_w, dinT, DM, DM, 0, 0);
    transpose_f32_to_bf16<<<dim3((DM * G4 + 255) / 256, 1, 1), 256, 0, stream>>>(
        k0, k0T, DM, G4, 0, 0);
    transpose_f32_to_bf16<<<dim3((DM * G4 + 255) / 256, 1, 1), 256, 0, stream>>>(
        r0, r0T, DM, G4, 0, 0);
    transpose_f32_to_bf16<<<dim3((DM * G4 + 255) / 256, 1, 1), 256, 0, stream>>>(
        k1, k1T, DM, G4, 0, 0);
    transpose_f32_to_bf16<<<dim3((DM * G4 + 255) / 256, 1, 1), 256, 0, stream>>>(
        r1, r1T, DM, G4, 0, 0);
  }

  // ---- embedding gather ----
  gather_embed<<<(MROWS * DM + 255) / 256, 256, 0, stream>>>(words, emb, y_bf, MROWS * DM);

  // ---- py = y @ P_w + P_b                     M=8192 N=512  K=512 ----
  gemm_bf16_wmma<<<dim3(DM / 256, MROWS / 64, 1), 256, 0, stream>>>(
      y_bf, 0, PwT, 0, nullptr, py_bf, 0, P_b, DM, DM);

  // ---- scores[b] = enc[b] @ py[b]^T           M=512  N=256  K=512 (x32) ----
  gemm_bf16_wmma<<<dim3(S_DEC / 256, S_ENC / 64, B_SZ), 256, 0, stream>>>(
      enc_bf, (i64)S_ENC * DM, py_bf, (i64)S_DEC * DM,
      scores, nullptr, (i64)S_ENC * S_DEC, nullptr, S_DEC, DM);

  // ---- mask + softmax over e, write a^T as bf16 ----
  softmax_cols<<<(B_SZ * S_DEC) / 8, 256, 0, stream>>>(scores, mask, at_bf);

  // ---- dec_in[b] = a^T[b] @ enc[b]            M=256  N=512  K=512 (x32) ----
  gemm_bf16_wmma<<<dim3(DM / 256, S_DEC / 64, B_SZ), 256, 0, stream>>>(
      at_bf, (i64)S_DEC * S_ENC, encT_bf, (i64)DM * S_ENC,
      nullptr, dec_bf, (i64)S_DEC * DM, nullptr, DM, S_ENC);

  // ---- x = dec_in @ din_w + din_b             M=8192 N=512  K=512 ----
  gemm_bf16_wmma<<<dim3(DM / 256, MROWS / 64, 1), 256, 0, stream>>>(
      dec_bf, 0, dinT, 0, nullptr, x_bf, 0, din_b, DM, DM);

  // ---- X0 = x @ k0 + b0 (hoisted input proj)  M=8192 N=2048 K=512 ----
  gemm_bf16_wmma<<<dim3(G4 / 256, MROWS / 64, 1), 256, 0, stream>>>(
      x_bf, 0, k0T, 0, X0, nullptr, 0, b0, G4, DM);

  // ---- LSTM recurrence: 256 steps, 2 layers ----
  lstm_init<<<(B_SZ * DM + 255) / 256, 256, 0, stream>>>(c0, c1, h0a, h1a);
  for (int t = 0; t < S_DEC; ++t) {
    const __bf16* h0_in  = (t & 1) ? h0b : h0a;
    __bf16*       h0_out = (t & 1) ? h0a : h0b;
    const __bf16* h1_in  = (t & 1) ? h1b : h1a;
    __bf16*       h1_out = (t & 1) ? h1a : h1b;
    lstm_layer0<<<DM / TJ, 256, 0, stream>>>(X0, t, h0_in, h0_out, c0, r0T);
    lstm_layer1<<<DM / TJ, 256, 0, stream>>>(t, h0_out, h1_in, h1_out, c1,
                                             k1T, r1T, b1, out);
  }
}
